// Attetion_20289425507095
// MI455X (gfx1250) — matrix operand order repaired and checked by reading, not compiled, not run on hardware
//
#include <hip/hip_runtime.h>

typedef __attribute__((ext_vector_type(16))) _Float16 v16h;
typedef __attribute__((ext_vector_type(8)))  _Float16 v8h;
typedef __attribute__((ext_vector_type(8)))  float    v8f;
typedef __attribute__((ext_vector_type(4)))  float    v4f;

#define WMMA_F16(a, b, c) \
  __builtin_amdgcn_wmma_f32_16x16x32_f16(false, (a), false, (b), (short)0, (c), false, false)

// CDNA5 async copy: 16B per lane, global -> LDS, tracked by ASYNCcnt.
// INST_OFFSET is added to BOTH the global and the LDS address (ISA 08 §4.4).
#define ASYNC_CP16(ldsoff, gptr, imm)                                   \
  asm volatile("global_load_async_to_lds_b128 %0, %1, off offset:%2"    \
               :: "v"(ldsoff), "v"(gptr), "i"(imm) : "memory")

#define WAIT_ASYNC(n) asm volatile("s_wait_asynccnt %0" :: "i"(n) : "memory")

constexpr int BB   = 4;
constexpr int SS   = 4096;
constexpr int DD   = 256;
constexpr int MTOK = BB * SS;   // 16384

// Generic pointer -> raw LDS byte offset (LDS aperture: addr[31:0] == offset).
__device__ __forceinline__ unsigned lds_off32(const void* p) {
  return (unsigned)(size_t)p;
}

// Assemble a 16-bit A-operand fragment from two contiguous 8-halfword chunks.
// A layout used here (16x32 f16): lane<16 holds K=0..15, lane>=16 holds K=16..31;
// caller passes pointers already offset by h*16 and h*16+8.
__device__ __forceinline__ v16h load_a16(const _Float16* p0, const _Float16* p1) {
  v8h x = *(const v8h*)p0;
  v8h y = *(const v8h*)p1;
  v16h r;
#pragma unroll
  for (int i = 0; i < 8; ++i) { r[i] = x[i]; r[8 + i] = y[i]; }
  return r;
}

// ---------------------------------------------------------------------------
// Kernel 1: Q/K/V projections.  Y[m,n] = sum_k x[m,k] * W[n,k]
// Each wave: 16 rows x 64 cols (4 accumulators). blockIdx.z selects Q/K/V.
// Q,K stored row-major f16 [MTOK][D]; V stored transposed f16 Vt[b][d][s].
// ---------------------------------------------------------------------------
__global__ __launch_bounds__(256) void qkv_proj_kernel(
    const float* __restrict__ x,  const float* __restrict__ Wq,
    const float* __restrict__ Wk, const float* __restrict__ Wv,
    _Float16* __restrict__ Qh, _Float16* __restrict__ Kh, _Float16* __restrict__ Vt)
{
  const int lane = threadIdx.x & 31;
  const int wave = threadIdx.x >> 5;
  const int h    = lane >> 4;      // lane half (0/1)
  const int ln   = lane & 15;
  const int pid  = blockIdx.z;     // 0=Q, 1=K, 2=V
  const float* __restrict__ W = (pid == 0) ? Wq : (pid == 1) ? Wk : Wv;

  const int mbase = (blockIdx.x * 8 + wave) * 16;   // 16 token rows per wave
  const int nbase = blockIdx.y * 64;                // 64 output cols per wave

  v8f acc[4];
#pragma unroll
  for (int j = 0; j < 4; ++j)
#pragma unroll
    for (int r = 0; r < 8; ++r) acc[j][r] = 0.0f;

  const float* __restrict__ xr = x + (size_t)(mbase + ln) * DD;

#pragma unroll
  for (int ks = 0; ks < 8; ++ks) {         // K-dim in steps of 32
    const int k0 = ks * 32 + h * 8;
    v16h a;
    {
      const v4f f0 = *(const v4f*)(xr + k0);
      const v4f f1 = *(const v4f*)(xr + k0 + 4);
      const v4f f2 = *(const v4f*)(xr + k0 + 16);
      const v4f f3 = *(const v4f*)(xr + k0 + 20);
#pragma unroll
      for (int i = 0; i < 4; ++i) {
        a[i]      = (_Float16)f0[i];
        a[4 + i]  = (_Float16)f1[i];
        a[8 + i]  = (_Float16)f2[i];
        a[12 + i] = (_Float16)f3[i];
      }
    }
    const int kh = ks * 32 + h * 8;
#pragma unroll
    for (int j = 0; j < 4; ++j) {
      const int n = nbase + j * 16 + ln;
      v16h b;
      const float* wr = W + (size_t)n * DD + kh;
      const v4f g0 = *(const v4f*)(wr);
      const v4f g1 = *(const v4f*)(wr + 4);
      const v4f g2 = *(const v4f*)(wr + 16);
      const v4f g3 = *(const v4f*)(wr + 20);
#pragma unroll
      for (int i = 0; i < 4; ++i) {
        b[i]      = (_Float16)g0[i];
        b[4 + i]  = (_Float16)g1[i];
        b[8 + i]  = (_Float16)g2[i];
        b[12 + i] = (_Float16)g3[i];
      }
      acc[j] = WMMA_F16(a, b, acc[j]);
    }
  }

  // C/D layout: c[r] = element(M = r + 8*h, N = ln)
  if (pid < 2) {
    _Float16* __restrict__ Out = (pid == 0) ? Qh : Kh;
#pragma unroll
    for (int j = 0; j < 4; ++j) {
      const int n = nbase + j * 16 + ln;
#pragma unroll
      for (int r = 0; r < 8; ++r) {
        const int m = mbase + r + 8 * h;
        Out[(size_t)m * DD + n] = (_Float16)acc[j][r];
      }
    }
  } else {
    // Transposed store: Vt[b][d][s]; per-lane contiguous in s -> one 16B store.
    const int bidx = mbase / SS;
    const int s0   = (mbase % SS) + 8 * h;
#pragma unroll
    for (int j = 0; j < 4; ++j) {
      const int d = nbase + j * 16 + ln;
      v8h vv;
#pragma unroll
      for (int r = 0; r < 8; ++r) vv[r] = (_Float16)acc[j][r];
      *(v8h*)(Vt + ((size_t)bidx * DD + d) * SS + s0) = vv;
    }
  }
}

// ---------------------------------------------------------------------------
// Kernel 2: flash attention with double-buffered async K/V staging in LDS.
// Block = 256 threads = 8 waves; each wave owns 16 queries (N dim), full D=256.
//   S^T = K * Q^T          (queries live in lanes -> per-lane softmax stats)
//   O^T += V^T * P^T       (16 d-tiles of f32 accumulators)
// Per 32-key block: K tile 16KB (flat contiguous) + Vt tile 16KB (64B per d-row),
// each thread issues 8x global_load_async_to_lds_b128.
// ---------------------------------------------------------------------------
__global__ __launch_bounds__(256) void flash_attn_kernel(
    const _Float16* __restrict__ Qh, const _Float16* __restrict__ Kh,
    const _Float16* __restrict__ Vt, float* __restrict__ out)
{
  __shared__ __align__(16) _Float16 ldsK[2][32 * DD];   // [key_local][d]
  __shared__ __align__(16) _Float16 ldsV[2][DD * 32];   // [d][key_local]

  const int tid  = threadIdx.x;
  const int lane = tid & 31;
  const int wave = tid >> 5;
  const int h    = lane >> 4;
  const int ln   = lane & 15;
  const int b    = blockIdx.y;
  const int q0   = blockIdx.x * 128 + wave * 16;
  const int q    = q0 + ln;                 // this lane's query (N column)
  const float scale = 0.0625f;              // 1/sqrt(256)

  const _Float16* __restrict__ Kb = Kh + (size_t)b * SS * DD;
  const _Float16* __restrict__ Vb = Vt + (size_t)b * DD * SS;

  // Preload Q^T B-fragments for all 8 K-steps: bq[ks][i] = Q[q, ks*32 + h*16 + i]
  v16h bq[8];
#pragma unroll
  for (int ks = 0; ks < 8; ++ks)
    bq[ks] = *(const v16h*)(Qh + (size_t)(b * SS + q) * DD + ks * 32 + h * 16);

  v8f acc[16];
#pragma unroll
  for (int t = 0; t < 16; ++t)
#pragma unroll
    for (int r = 0; r < 8; ++r) acc[t][r] = 0.0f;

  float mrow = -__builtin_inff();
  float lrow = 0.0f;

  // ---- async stage of one 32-key block (8 x 16B per thread) ----
  auto stage = [&](int buf, int key0) {
    // K tile: 16KB contiguous global range starting at row key0
    {
      const unsigned loff = lds_off32(&ldsK[buf][0]) + tid * 16;
      const char* g = (const char*)(Kb + (size_t)key0 * DD) + tid * 16;
      ASYNC_CP16(loff, g, 0);
      ASYNC_CP16(loff, g, 4096);
      ASYNC_CP16(loff, g, 8192);
      ASYNC_CP16(loff, g, 12288);
    }
    // Vt tile: thread t copies d-row t (64B) -> ldsV[buf][t*32 ..]
    {
      const unsigned loff = lds_off32(&ldsV[buf][0]) + tid * 64;
      const char* g = (const char*)(Vb + (size_t)tid * SS + key0);
      ASYNC_CP16(loff, g, 0);
      ASYNC_CP16(loff, g, 16);
      ASYNC_CP16(loff, g, 32);
      ASYNC_CP16(loff, g, 48);
    }
  };

  constexpr int NBLK = SS / 32;   // 128
  stage(0, 0);

  for (int i = 0; i < NBLK; ++i) {
    const int cur = i & 1;
    if (i + 1 < NBLK) {           // issue next block's prefetch, then wait for cur
      stage(cur ^ 1, (i + 1) * 32);
      WAIT_ASYNC(8);              // <=8 outstanding => current buffer complete
    } else {
      WAIT_ASYNC(0);
    }
    __syncthreads();              // cur tiles visible to all waves

    const _Float16* __restrict__ kt = &ldsK[cur][0];
    const _Float16* __restrict__ vt = &ldsV[cur][0];

    // --- S^T = K * Q^T : two 16x16 f32 fragments (keys 0..15 / 16..31 of block)
    v8f sc0, sc1;
#pragma unroll
    for (int r = 0; r < 8; ++r) { sc0[r] = 0.0f; sc1[r] = 0.0f; }
#pragma unroll
    for (int ks = 0; ks < 8; ++ks) {
      const _Float16* kp0 = kt + (size_t)ln * DD + ks * 32 + h * 16;
      sc0 = WMMA_F16(load_a16(kp0, kp0 + 8), bq[ks], sc0);
      const _Float16* kp1 = kt + (size_t)(16 + ln) * DD + ks * 32 + h * 16;
      sc1 = WMMA_F16(load_a16(kp1, kp1 + 8), bq[ks], sc1);
    }

    // --- online softmax (query = ln; keys split across the two lane-halves)
    float p0[8], p1[8];
    float bmax = -__builtin_inff();
#pragma unroll
    for (int r = 0; r < 8; ++r) {
      p0[r] = sc0[r] * scale;
      p1[r] = sc1[r] * scale;
      bmax  = fmaxf(bmax, fmaxf(p0[r], p1[r]));
    }
    bmax = fmaxf(bmax, __shfl_xor(bmax, 16, 32));
    const float mnew = fmaxf(mrow, bmax);
    const float corr = __expf(mrow - mnew);
    float rsum = 0.0f;
#pragma unroll
    for (int r = 0; r < 8; ++r) {
      p0[r] = __expf(p0[r] - mnew);
      p1[r] = __expf(p1[r] - mnew);
      rsum += p0[r] + p1[r];
    }
    rsum += __shfl_xor(rsum, 16, 32);
    lrow = lrow * corr + rsum;
    mrow = mnew;

    // rescale output accumulators
#pragma unroll
    for (int t = 0; t < 16; ++t)
#pragma unroll
      for (int r = 0; r < 8; ++r) acc[t][r] *= corr;

    // --- build P^T B-fragment (32 keys x 16 queries): b[i]=(K=h*16+i, N=ln)
    v16h pb;
#pragma unroll
    for (int r = 0; r < 8; ++r) {
      const float o0 = __shfl_xor(p0[r], 16, 32);
      const float o1 = __shfl_xor(p1[r], 16, 32);
      pb[r]     = (_Float16)(h ? o1 : p0[r]);   // keys h*16 + r
      pb[8 + r] = (_Float16)(h ? p1[r] : o0);   // keys h*16 + 8 + r
    }

    // --- O^T += V^T * P^T  over 16 d-tiles
#pragma unroll
    for (int t = 0; t < 16; ++t) {
      const int d = t * 16 + ln;
      const _Float16* vp = vt + (size_t)d * 32 + h * 16;
      acc[t] = WMMA_F16(load_a16(vp, vp + 8), pb, acc[t]);
    }

    __syncthreads();              // all reads of cur done before it is re-staged
  }

  // --- epilogue: out[(b*S+q)*D + d] = acc / l ; d = t*16 + 8h + r (contiguous)
  const float inv_l = 1.0f / lrow;
  float* __restrict__ op = out + (size_t)(b * SS + q) * DD;
#pragma unroll
  for (int t = 0; t < 16; ++t) {
    v4f o0, o1;
#pragma unroll
    for (int r = 0; r < 4; ++r) {
      o0[r] = acc[t][r] * inv_l;
      o1[r] = acc[t][4 + r] * inv_l;
    }
    *(v4f*)(op + t * 16 + 8 * h)     = o0;
    *(v4f*)(op + t * 16 + 8 * h + 4) = o1;
  }
}

extern "C" void kernel_launch(void* const* d_in, const int* in_sizes, int n_in,
                              void* d_out, int out_size, void* d_ws, size_t ws_size,
                              hipStream_t stream) {
  const float* x  = (const float*)d_in[0];
  const float* Wq = (const float*)d_in[1];
  const float* Wk = (const float*)d_in[2];
  const float* Wv = (const float*)d_in[3];
  float* out = (float*)d_out;

  _Float16* Qh = (_Float16*)d_ws;
  _Float16* Kh = Qh + (size_t)MTOK * DD;
  _Float16* Vt = Kh + (size_t)MTOK * DD;

  dim3 pgrid(MTOK / 128, DD / 64, 3);   // 128 rows x 64 cols per block, z = Q/K/V
  qkv_proj_kernel<<<pgrid, 256, 0, stream>>>(x, Wq, Wk, Wv, Qh, Kh, Vt);

  dim3 agrid(SS / 128, BB);             // 128 queries per block, y = batch
  flash_attn_kernel<<<agrid, 256, 0, stream>>>(Qh, Kh, Vt, out);
}
